// MoE_13941463843216
// MI455X (gfx1250) — matrix operand order repaired and checked by reading, not compiled, not run on hardware
//
#include <hip/hip_runtime.h>
#include <hip/hip_bf16.h>

#define EMBED 64
#define NEXP  4
#define FFN   128
#define NTOK  262144
#define NTILE (NTOK / 16)
#define BLOCKS 1024
#define THREADS 256

typedef __attribute__((ext_vector_type(16))) __bf16 v16bf;
typedef __attribute__((ext_vector_type(8)))  float  v8f;

union Frag {
    v16bf v;
    uint4 q[2];
    unsigned int u[8];
    __bf16 s[16];
};

__device__ __forceinline__ unsigned pk_bf16(float a, float b) {
    __bf16 x = (__bf16)a, y = (__bf16)b;
    unsigned short ux = __builtin_bit_cast(unsigned short, x);
    unsigned short uy = __builtin_bit_cast(unsigned short, y);
    return (unsigned)ux | ((unsigned)uy << 16);
}

__device__ __forceinline__ v8f load8f(const float* p) {
    float4 a = ((const float4*)p)[0];
    float4 b = ((const float4*)p)[1];
    v8f r;
    r[0]=a.x; r[1]=a.y; r[2]=a.z; r[3]=a.w;
    r[4]=b.x; r[5]=b.y; r[6]=b.z; r[7]=b.w;
    return r;
}

__device__ __forceinline__ Frag loadA(const __bf16* base) {
    // A-frag 16x32 bf16: lane (m=lane&15, h=lane>>4) holds
    // K = {16h..16h+15} -> one contiguous 32B run.
    Frag a;
    a.q[0] = *(const uint4*)(base);
    a.q[1] = *(const uint4*)(base + 8);
    return a;
}

__launch_bounds__(THREADS, 1)
__global__ void moe_top2_wmma(const float* __restrict__ x,
                              const float* __restrict__ Wg,
                              const float* __restrict__ W1,
                              const float* __restrict__ b1,
                              const float* __restrict__ W2,
                              const float* __restrict__ b2,
                              float* __restrict__ out) {
    extern __shared__ unsigned char smem[];
    __bf16* sW1 = (__bf16*)smem;                 // [E][F][D]  bf16, 64KB
    __bf16* sW2 = sW1 + NEXP * FFN * EMBED;      // [E][D][F]  bf16, 64KB
    __bf16* sWg = sW2 + NEXP * EMBED * FFN;      // [16][D]    bf16 (rows 4..15 = 0), 2KB
    float*  sb1 = (float*)(sWg + 16 * EMBED);    // [E][F]     f32, 2KB
    float*  sb2 = sb1 + NEXP * FFN;              // [E][D]     f32, 1KB

    const int tid = threadIdx.x;
    // ---- stage weights to LDS (bf16) ----
    for (int i = tid; i < NEXP * FFN * EMBED; i += THREADS) sW1[i] = (__bf16)W1[i];
    for (int i = tid; i < NEXP * EMBED * FFN; i += THREADS) sW2[i] = (__bf16)W2[i];
    for (int i = tid; i < 16 * EMBED; i += THREADS)
        sWg[i] = (i < NEXP * EMBED) ? (__bf16)Wg[i] : (__bf16)0.0f;
    for (int i = tid; i < NEXP * FFN; i += THREADS) sb1[i] = b1[i];
    for (int i = tid; i < NEXP * EMBED; i += THREADS) sb2[i] = b2[i];
    __syncthreads();

    const int lane = tid & 31;
    const int n = lane & 15;   // token within tile / WMMA N index
    const int h = lane >> 4;   // lane half
    const int wave = blockIdx.x * (THREADS / 32) + (tid >> 5);
    const int nwaves = gridDim.x * (THREADS / 32);

    #pragma unroll 1
    for (int tile = wave; tile < NTILE; tile += nwaves) {
        const int tb = tile * 16;

        // prefetch next tile's x rows (speculative, gfx1250 global_prefetch)
        {
            int ntile = tile + nwaves;
            if (ntile < NTILE) {
                const float* np = x + (size_t)(ntile * 16 + n) * EMBED + 16 * h;
                __builtin_prefetch(np, 0, 3);
                __builtin_prefetch(np + 32, 0, 3);
            }
        }

        // ---- x^T B-fragments: 2 K-tiles of 32 (K = embed dim) ----
        // B layout: lane (n, h), element e = x[tb+n][32*kt + 16*h + e]
        Frag xB[2];
        #pragma unroll
        for (int kt = 0; kt < 2; ++kt) {
            const float* p = x + (size_t)(tb + n) * EMBED + 32 * kt + 16 * h;
            #pragma unroll
            for (int j = 0; j < 4; ++j) {
                float4 f = ((const float4*)p)[j];
                xB[kt].s[4 * j + 0] = (__bf16)f.x;
                xB[kt].s[4 * j + 1] = (__bf16)f.y;
                xB[kt].s[4 * j + 2] = (__bf16)f.z;
                xB[kt].s[4 * j + 3] = (__bf16)f.w;
            }
        }

        // ---- gating: logits^T (16-pad x 16) = Wg_pad (16x64) @ x^T ----
        v8f accG;
        #pragma unroll
        for (int r = 0; r < 8; ++r) accG[r] = 0.0f;
        #pragma unroll
        for (int kt = 0; kt < 2; ++kt) {
            Frag a = loadA(sWg + n * EMBED + 32 * kt + 16 * h);
            accG = __builtin_amdgcn_wmma_f32_16x16x32_bf16(
                false, a.v, false, xB[kt].v, (short)0, accG, false, false);
        }
        // lanes h==0 hold logits[token n][e] in accG[e], e=0..3 (rows 8..15 are zero pad).
        // Branchless top-2-of-4 (sorting network, ties prefer lower index), then
        // softmax over the two selected logits.
        float w0, w1, w2, w3;
        {
            float l0 = accG[0], l1 = accG[1], l2 = accG[2], l3 = accG[3];
            bool b01 = l1 > l0;
            float hi01 = b01 ? l1 : l0, lo01 = b01 ? l0 : l1;
            int   ih01 = b01 ? 1 : 0;
            bool b23 = l3 > l2;
            float hi23 = b23 ? l3 : l2, lo23 = b23 ? l2 : l3;
            int   ih23 = b23 ? 3 : 2;
            bool bw = hi23 > hi01;                       // winner pair is {2,3}?
            int   i0 = bw ? ih23 : ih01;
            float m0 = bw ? hi23 : hi01;
            float cA = bw ? lo23 : lo01;                 // loser of winning pair
            float cB = bw ? hi01 : hi23;                 // winner of losing pair
            int   iA = bw ? (b23 ? 2 : 3) : (b01 ? 0 : 1);
            int   iB = bw ? ih01 : ih23;
            // tie pref: lower index. bw -> cB has lower indices; !bw -> cA does.
            bool pickA = bw ? (cA > cB) : (cA >= cB);
            float m1 = pickA ? cA : cB;
            int   i1 = pickA ? iA : iB;

            float e1 = __expf(m1 - m0);
            // fast reciprocal (v_rcp_f32) instead of IEEE divide sequence
            float inv = __builtin_amdgcn_rcpf(1.0f + e1);
            float p0 = inv, p1 = e1 * inv;
            w0 = (i0 == 0) ? p0 : ((i1 == 0) ? p1 : 0.0f);
            w1 = (i0 == 1) ? p0 : ((i1 == 1) ? p1 : 0.0f);
            w2 = (i0 == 2) ? p0 : ((i1 == 2) ? p1 : 0.0f);
            w3 = (i0 == 3) ? p0 : ((i1 == 3) ? p1 : 0.0f);
            // broadcast to h==1 lanes (same token n lives at lane^16)
            float t0 = __shfl_xor(w0, 16, 32);
            float t1 = __shfl_xor(w1, 16, 32);
            float t2 = __shfl_xor(w2, 16, 32);
            float t3 = __shfl_xor(w3, 16, 32);
            w0 = h ? t0 : w0;
            w1 = h ? t1 : w1;
            w2 = h ? t2 : w2;
            w3 = h ? t3 : w3;
        }
        const float w[NEXP] = {w0, w1, w2, w3};

        v8f accOut[4];
        #pragma unroll
        for (int at = 0; at < 4; ++at)
            #pragma unroll
            for (int r = 0; r < 8; ++r) accOut[at][r] = 0.0f;

        #pragma unroll 1
        for (int e = 0; e < NEXP; ++e) {
            // ---- layer 1: h^T (128x16) = W1_e (128x64) @ x^T, C init = b1 ----
            // accH[mt]: lane (n,h) holds h^T[16*mt + 8*h + r][n]
            v8f accH[8];
            #pragma unroll
            for (int mt = 0; mt < 8; ++mt)
                accH[mt] = load8f(sb1 + e * FFN + 16 * mt + 8 * h);
            #pragma unroll
            for (int kt = 0; kt < 2; ++kt) {
                #pragma unroll
                for (int mt = 0; mt < 8; ++mt) {
                    Frag a = loadA(sW1 + (size_t)(e * FFN + 16 * mt + n) * EMBED
                                   + 32 * kt + 16 * h);
                    accH[mt] = __builtin_amdgcn_wmma_f32_16x16x32_bf16(
                        false, a.v, false, xB[kt].v, (short)0, accH[mt], false, false);
                }
            }
            // relu + pack to bf16 pairs
            unsigned hp[8][4];
            #pragma unroll
            for (int mt = 0; mt < 8; ++mt)
                #pragma unroll
                for (int j = 0; j < 4; ++j)
                    hp[mt][j] = pk_bf16(fmaxf(accH[mt][2 * j], 0.0f),
                                        fmaxf(accH[mt][2 * j + 1], 0.0f));

            // ---- layer 2: y^T (64x16) = W2_e (64x128) @ h^T, C init = b2 ----
            v8f accY[4];
            #pragma unroll
            for (int at = 0; at < 4; ++at)
                accY[at] = load8f(sb2 + e * EMBED + 16 * at + 8 * h);
            #pragma unroll
            for (int kt = 0; kt < 4; ++kt) {
                // Build h^T B-fragment for K-tile kt from C/D-layout accH:
                // need F = 32*kt + 16*h + e; own half supplies e<8 (h=0) / e>=8 (h=1),
                // partner lane (lane^16) supplies the rest — pure shfl_xor, no LDS.
                Frag bh;
                #pragma unroll
                for (int j = 0; j < 4; ++j) {
                    unsigned own_lo = hp[2 * kt + 0][j];
                    unsigned own_hi = hp[2 * kt + 1][j];
                    unsigned t0 = __shfl_xor(own_lo, 16, 32); // partner's hp[2kt+0]
                    unsigned t1 = __shfl_xor(own_hi, 16, 32); // partner's hp[2kt+1]
                    bh.u[j]     = h ? t1 : own_lo;  // elements e = 0..7
                    bh.u[4 + j] = h ? own_hi : t0;  // elements e = 8..15
                }
                #pragma unroll
                for (int at = 0; at < 4; ++at) {
                    Frag a = loadA(sW2 + (size_t)(e * EMBED + 16 * at + n) * FFN
                                   + 32 * kt + 16 * h);
                    accY[at] = __builtin_amdgcn_wmma_f32_16x16x32_bf16(
                        false, a.v, false, bh.v, (short)0, accY[at], false, false);
                }
            }
            // ---- weighted combine: token index == lane N index -> scalar weight ----
            const float we = w[e];
            #pragma unroll
            for (int at = 0; at < 4; ++at)
                #pragma unroll
                for (int r = 0; r < 8; ++r)
                    accOut[at][r] += we * accY[at][r];
        }

        // ---- store: lane (n,h) owns out[tb+n][16*at + 8*h + r] ----
        #pragma unroll
        for (int at = 0; at < 4; ++at) {
            float* op = out + (size_t)(tb + n) * EMBED + 16 * at + 8 * h;
            float4 o0 = {accOut[at][0], accOut[at][1], accOut[at][2], accOut[at][3]};
            float4 o1 = {accOut[at][4], accOut[at][5], accOut[at][6], accOut[at][7]};
            ((float4*)op)[0] = o0;
            ((float4*)op)[1] = o1;
        }
    }
}

extern "C" void kernel_launch(void* const* d_in, const int* in_sizes, int n_in,
                              void* d_out, int out_size, void* d_ws, size_t ws_size,
                              hipStream_t stream) {
    (void)in_sizes; (void)n_in; (void)out_size; (void)d_ws; (void)ws_size;
    const float* x  = (const float*)d_in[0];
    const float* Wg = (const float*)d_in[1];
    const float* W1 = (const float*)d_in[2];
    const float* b1 = (const float*)d_in[3];
    const float* W2 = (const float*)d_in[4];
    const float* b2 = (const float*)d_in[5];
    float* out = (float*)d_out;

    const size_t smem = (size_t)(NEXP * FFN * EMBED) * 2   // sW1 bf16
                      + (size_t)(NEXP * EMBED * FFN) * 2   // sW2 bf16
                      + (size_t)(16 * EMBED) * 2           // sWg bf16 (padded)
                      + (size_t)(NEXP * FFN) * 4           // sb1 f32
                      + (size_t)(NEXP * EMBED) * 4;        // sb2 f32

    moe_top2_wmma<<<BLOCKS, THREADS, smem, stream>>>(x, Wg, W1, b1, W2, b2, out);
}
